// GRUD_36799279792833
// MI455X (gfx1250) — compile-verified, matrix operand order checked by reading, not run on hardware
//
#include <hip/hip_runtime.h>
#include <hip/hip_bf16.h>

typedef __attribute__((ext_vector_type(16))) _Float16 v16h;
typedef __attribute__((ext_vector_type(8)))  float    v8f;

#define NB     4096
#define NT     128
#define NIN    128
#define NH     32
#define NE     128
#define NCODES 8000

// ---------------------------------------------------------------------------
// CDNA5 WMMA fragment helpers (wave32, 16x16x32 f16 -> f32)
// A (16x32 MxK): lanes 0-15 hold M=lane, VGPR j covers K=(j/4)*16+(j%4)*2(+1);
//                lanes 16-31 same M, K += 8.
// B (32x16 KxN): N = lane%16, K pattern mirrors A.
// C/D (16x16 f32, 8 VGPRs): VGPR r -> M = r + 8*(lane/16), N = lane%16.
// ---------------------------------------------------------------------------

__device__ __forceinline__ v8f wmma_f16(v16h a, v16h b, v8f c) {
  return __builtin_amdgcn_wmma_f32_16x16x32_f16(false, a, false, b,
                                                (short)0, c, false, false);
}

__device__ __forceinline__ v16h afrag_f32(const float* src, int ld, int kbase) {
  const int lane = threadIdx.x & 31;
  const int half = lane >> 4;
  const int m    = lane & 15;
  v16h a;
#pragma unroll
  for (int j = 0; j < 8; ++j) {
    int k = kbase + ((j >> 2) << 4) + (half << 3) + ((j & 3) << 1);
    float2 v = *(const float2*)(src + (size_t)m * ld + k);
    a[2 * j]     = (_Float16)v.x;
    a[2 * j + 1] = (_Float16)v.y;
  }
  return a;
}

// B-fragment built from row-major f32 weights W[K][N] (pack-time only)
__device__ __forceinline__ v16h bfrag_f32(const float* W, int ld, int kbase,
                                          int nbase) {
  const int lane = threadIdx.x & 31;
  const int half = lane >> 4;
  const int n    = nbase + (lane & 15);
  v16h b;
#pragma unroll
  for (int j = 0; j < 8; ++j) {
    int k = kbase + ((j >> 2) << 4) + (half << 3) + ((j & 3) << 1);
    b[2 * j]     = (_Float16)W[(size_t)k * ld + n];
    b[2 * j + 1] = (_Float16)W[(size_t)(k + 1) * ld + n];
  }
  return b;
}

// Pre-packed B-fragment: frag f occupies 512 f16 (32 B per lane, contiguous)
__device__ __forceinline__ v16h bfrag_packed(const _Float16* P, int frag) {
  const int lane = threadIdx.x & 31;
  return *(const v16h*)(P + (size_t)frag * 512 + lane * 16);
}

// ---------------------------------------------------------------------------
// Async copy (GLOBAL_LOAD_ASYNC_TO_LDS_B128) of a [16 x COLS] f32 tile into
// LDS. ASYNCcnt += COLS/8 per call.
// ---------------------------------------------------------------------------
template <int COLS>
__device__ __forceinline__ void async_tile_load(const float* gsrc,
                                                size_t grstride,
                                                float* lds_tile) {
  const int lane = threadIdx.x & 31;
  const int row  = lane >> 1;
  const int c0   = (lane & 1) * (COLS / 2);
  unsigned lofs = (unsigned)(size_t)(lds_tile + row * COLS + c0);
  const float* g = gsrc + (size_t)row * grstride + c0;
#pragma unroll
  for (int i = 0; i < COLS / 8; ++i) {
    asm volatile("global_load_async_to_lds_b128 %0, %1, off"
                 :: "v"(lofs + (unsigned)(i * 16)),
                    "v"((unsigned long long)(size_t)(g + i * 4))
                 : "memory");
  }
}

__device__ __forceinline__ void wait_async_le16() {
  asm volatile("s_wait_asynccnt 16" ::: "memory");
}
__device__ __forceinline__ void wait_async_le4() {
  asm volatile("s_wait_asynccnt 4" ::: "memory");
}
__device__ __forceinline__ void wait_async_0() {
  asm volatile("s_wait_asynccnt 0" ::: "memory");
}
__device__ __forceinline__ void wait_ds0() {
  asm volatile("s_wait_dscnt 0" ::: "memory");
}

// ---------------------------------------------------------------------------
// Tensor Data Mover: single-descriptor load of a 16x128 f32 tile into LDS.
// D# built per CDNA5 ISA ch.8: group0 {count=1, lds_addr, global_addr,
// type=2}; group1 {data_size=4B, tensor_dim0=128, tensor_dim1=16,
// tile_dim0=128, tile_dim1=16, tensor_dim0_stride=NT*NIN}; groups 2/3 zero
// (tile_dim2=0 -> 2D tile). Tracked by TENSORcnt.
// ---------------------------------------------------------------------------
#if __has_builtin(__builtin_amdgcn_tensor_load_to_lds) && \
    __has_builtin(__builtin_amdgcn_s_wait_tensorcnt)
#define HAVE_TDM 1
typedef unsigned int u32x4 __attribute__((ext_vector_type(4)));
typedef int          i32x4 __attribute__((ext_vector_type(4)));
typedef int          i32x8 __attribute__((ext_vector_type(8)));

__device__ __forceinline__ void tdm_load_tile_16x128_f32(const float* gsrc,
                                                         float* lds_tile) {
  unsigned long long ga = (unsigned long long)(size_t)gsrc;
  u32x4 g0;
  g0[0] = 1u;                                    // count=1, user D#
  g0[1] = (unsigned)(size_t)lds_tile;            // lds_addr (bytes)
  g0[2] = (unsigned)(ga & 0xFFFFFFFFull);        // global_addr[31:0]
  g0[3] = (unsigned)((ga >> 32) & 0x01FFFFFFull) // global_addr[56:32]
          | 0x80000000u;                         // type=2 ("image")
  i32x8 g1;
  g1[0] = 0x00020000;      // wg_mask=0 | data_size=2 (4 B) | flags=0
  g1[1] = NIN << 16;       // atomic_barrier_addr=0 | tensor_dim0 = 128
  g1[2] = 16 << 16;        // tensor_dim0 hi=0 | tensor_dim1 = 16
  g1[3] = NIN << 16;       // tensor_dim1 hi=0 | tile_dim0 = 128
  g1[4] = 16;              // tile_dim1 = 16 | tile_dim2 = 0 (2D tile)
  g1[5] = NT * NIN;        // tensor_dim0_stride = 16384 elems
  g1[6] = 0;               // stride hi | tensor_dim1_stride lo
  g1[7] = 0;
  i32x4 gz = {0, 0, 0, 0};
#if __clang_major__ >= 23
  i32x8 gz8 = {0, 0, 0, 0, 0, 0, 0, 0};
  __builtin_amdgcn_tensor_load_to_lds(g0, g1, gz, gz, gz8, 0);
#else
  __builtin_amdgcn_tensor_load_to_lds(g0, g1, gz, gz, 0);
#endif
}

__device__ __forceinline__ void grud_tile_load(const float* gsrc,
                                               float* lds_tile) {
  tdm_load_tile_16x128_f32(gsrc, lds_tile);
}
__device__ __forceinline__ void grud_wait_prev() {
  __builtin_amdgcn_s_wait_tensorcnt((short)1);  // older TDM op complete
}
__device__ __forceinline__ void grud_wait_all() {
  __builtin_amdgcn_s_wait_tensorcnt((short)0);
}
#else
__device__ __forceinline__ void grud_tile_load(const float* gsrc,
                                               float* lds_tile) {
  async_tile_load<NIN>(gsrc, (size_t)NT * NIN, lds_tile);
}
__device__ __forceinline__ void grud_wait_prev() { wait_async_le16(); }
__device__ __forceinline__ void grud_wait_all() { wait_async_0(); }
#endif

// ---------------------------------------------------------------------------
// Pack kernels: W[K][N] f32 -> fragment-contiguous f16 (frag = kb*(N/16)+nb)
// ---------------------------------------------------------------------------
__global__ __launch_bounds__(32) void pack_weight_kernel(
    const float* __restrict__ W, int ld, int nNB, _Float16* __restrict__ P) {
  const int f  = blockIdx.x;
  const int kb = f / nNB, nb = f % nNB;
  v16h b = bfrag_f32(W, ld, kb * 32, nb * 16);
  *(v16h*)(P + (size_t)f * 512 + (threadIdx.x & 31) * 16) = b;
}

__global__ __launch_bounds__(32) void pack_embed_kernel(
    const float* __restrict__ embedAll, const int* __restrict__ mapInfo,
    _Float16* __restrict__ P) {
  const int f    = blockIdx.x;  // kb*8 + nb
  const int kb   = f >> 3, nb = f & 7;
  const int lane = threadIdx.x & 31;
  const int half = lane >> 4;
  const int n    = nb * 16 + (lane & 15);
  v16h b;
#pragma unroll
  for (int j = 0; j < 8; ++j) {
    int k = kb * 32 + ((j >> 2) << 4) + (half << 3) + ((j & 3) << 1);
    b[2 * j]     = (_Float16)embedAll[(size_t)mapInfo[k] * NE + n];
    b[2 * j + 1] = (_Float16)embedAll[(size_t)mapInfo[k + 1] * NE + n];
  }
  *(v16h*)(P + (size_t)f * 512 + lane * 16) = b;
}

// ---------------------------------------------------------------------------
// Kernel 1: GRU-D scan. One wave owns 16 batch rows; h state in C-fragment
// registers across 128 timesteps. Delta tiles stream via TDM (or async)
// double buffering; weights from packed B-fragments; LDS relayout for x, r*h.
// ---------------------------------------------------------------------------
__global__ __launch_bounds__(128) void grud_scan_kernel(
    const float* __restrict__ X, const float* __restrict__ Mask,
    const float* __restrict__ Delta, const float* __restrict__ Xlast,
    const int* __restrict__ seq_len, const float* __restrict__ X_mean,
    const float* __restrict__ b_z, const float* __restrict__ b_r,
    const float* __restrict__ b_h, const float* __restrict__ b_gx,
    const float* __restrict__ b_gh, const _Float16* __restrict__ pW_z,
    const _Float16* __restrict__ pW_r, const _Float16* __restrict__ pW_h,
    const _Float16* __restrict__ pW_gx, const _Float16* __restrict__ pW_gh,
    float* __restrict__ clinic) {
  __shared__ float dstage[4][2][16][NIN];  // TDM/async double buffer (64 KB)
  __shared__ float xstage[4][16][NIN];     // C->A relayout (32 KB)
  __shared__ float hstage[4][16][NH];      // C->A relayout (8 KB)

  const int wave = threadIdx.x >> 5;
  const int lane = threadIdx.x & 31;
  const int half = lane >> 4;
  const int ncol = lane & 15;
  const int b0   = (blockIdx.x * 4 + wave) * 16;

  float(*xs)[NIN] = xstage[wave];
  float(*hs)[NH]  = hstage[wave];

  v8f hc[2] = {};
  v8f zero  = {};

  int sl[8];
#pragma unroll
  for (int r = 0; r < 8; ++r) sl[r] = seq_len[b0 + r + 8 * half];

  float bgx_l[8], xmean_l[8];
#pragma unroll
  for (int nt = 0; nt < 8; ++nt) {
    bgx_l[nt]   = b_gx[nt * 16 + ncol];
    xmean_l[nt] = X_mean[nt * 16 + ncol];
  }
  float bgh_l[2], bz_l[2], br_l[2], bh_l[2];
#pragma unroll
  for (int nt = 0; nt < 2; ++nt) {
    bgh_l[nt] = b_gh[nt * 16 + ncol];
    bz_l[nt]  = b_z[nt * 16 + ncol];
    br_l[nt]  = b_r[nt * 16 + ncol];
    bh_l[nt]  = b_h[nt * 16 + ncol];
  }

  const size_t rowstride = (size_t)NT * NIN;

  grud_tile_load(Delta + (size_t)b0 * rowstride, &dstage[wave][0][0][0]);

  for (int t = 0; t < NT; ++t) {
    if (t + 1 < NT) {
      grud_tile_load(Delta + (size_t)b0 * rowstride + (t + 1) * NIN,
                     &dstage[wave][(t + 1) & 1][0][0]);
      grud_wait_prev();  // tile t complete; tile t+1 may be in flight
    } else {
      grud_wait_all();
    }

    const float* dtile = &dstage[wave][t & 1][0][0];
    v16h dA[4];
#pragma unroll
    for (int kk = 0; kk < 4; ++kk) dA[kk] = afrag_f32(dtile, NIN, kk * 32);

    // ---- input decay + imputation -> x stage
#pragma unroll
    for (int nt = 0; nt < 8; ++nt) {
      v8f c = zero;
#pragma unroll
      for (int kk = 0; kk < 4; ++kk)
        c = wmma_f16(dA[kk], bfrag_packed(pW_gx, kk * 8 + nt), c);
#pragma unroll
      for (int r = 0; r < 8; ++r) {
        int row   = r + 8 * half;
        int col   = nt * 16 + ncol;
        float dx  = __expf(-fmaxf(c[r] + bgx_l[nt], 0.f));
        size_t gi = ((size_t)(b0 + row) * NT + t) * NIN + col;
        float mk = Mask[gi], xv = X[gi], xl = Xlast[gi];
        xs[row][col] =
            mk * xv + (1.f - mk) * (dx * xl + (1.f - dx) * xmean_l[nt]);
      }
    }

    // ---- hidden decay: h = exp(-relu(d@W_gh+b)) * h ; stage h
#pragma unroll
    for (int nt = 0; nt < 2; ++nt) {
      v8f c = zero;
#pragma unroll
      for (int kk = 0; kk < 4; ++kk)
        c = wmma_f16(dA[kk], bfrag_packed(pW_gh, kk * 2 + nt), c);
#pragma unroll
      for (int r = 0; r < 8; ++r) {
        float dh  = __expf(-fmaxf(c[r] + bgh_l[nt], 0.f));
        float hv  = dh * hc[nt][r];
        hc[nt][r] = hv;
        hs[r + 8 * half][nt * 16 + ncol] = hv;
      }
    }

    wait_ds0();

    v16h xA[4];
#pragma unroll
    for (int kk = 0; kk < 4; ++kk) xA[kk] = afrag_f32(&xs[0][0], NIN, kk * 32);
    v16h hA = afrag_f32(&hs[0][0], NH, 0);
    const float* mbase = Mask + (size_t)b0 * rowstride + t * NIN;
    v16h mA[4];
#pragma unroll
    for (int kk = 0; kk < 4; ++kk)
      mA[kk] = afrag_f32(mbase, (int)rowstride, kk * 32);

    // ---- z and r gates: comb[16,288] @ W[288,32]
    v8f zt[2], rt[2];
#pragma unroll
    for (int nt = 0; nt < 2; ++nt) {
      v8f cz = zero, cr = zero;
#pragma unroll
      for (int kk = 0; kk < 4; ++kk) {
        cz = wmma_f16(xA[kk], bfrag_packed(pW_z, kk * 2 + nt), cz);
        cr = wmma_f16(xA[kk], bfrag_packed(pW_r, kk * 2 + nt), cr);
      }
      cz = wmma_f16(hA, bfrag_packed(pW_z, 8 + nt), cz);
      cr = wmma_f16(hA, bfrag_packed(pW_r, 8 + nt), cr);
#pragma unroll
      for (int kk = 0; kk < 4; ++kk) {
        cz = wmma_f16(mA[kk], bfrag_packed(pW_z, (5 + kk) * 2 + nt), cz);
        cr = wmma_f16(mA[kk], bfrag_packed(pW_r, (5 + kk) * 2 + nt), cr);
      }
#pragma unroll
      for (int r = 0; r < 8; ++r) {
        zt[nt][r] = 1.f / (1.f + __expf(-(cz[r] + bz_l[nt])));
        rt[nt][r] = 1.f / (1.f + __expf(-(cr[r] + br_l[nt])));
      }
    }

    // ---- stage r*h, reload as A-fragment
#pragma unroll
    for (int nt = 0; nt < 2; ++nt)
#pragma unroll
      for (int r = 0; r < 8; ++r)
        hs[r + 8 * half][nt * 16 + ncol] = rt[nt][r] * hc[nt][r];
    wait_ds0();
    v16h rhA = afrag_f32(&hs[0][0], NH, 0);

    // ---- h_tilde, state update, clinic capture at t+1 == seq_len
#pragma unroll
    for (int nt = 0; nt < 2; ++nt) {
      v8f ch = zero;
#pragma unroll
      for (int kk = 0; kk < 4; ++kk)
        ch = wmma_f16(xA[kk], bfrag_packed(pW_h, kk * 2 + nt), ch);
      ch = wmma_f16(rhA, bfrag_packed(pW_h, 8 + nt), ch);
#pragma unroll
      for (int kk = 0; kk < 4; ++kk)
        ch = wmma_f16(mA[kk], bfrag_packed(pW_h, (5 + kk) * 2 + nt), ch);
#pragma unroll
      for (int r = 0; r < 8; ++r) {
        float ht  = tanhf(ch[r] + bh_l[nt]);
        float z   = zt[nt][r];
        float hv  = (1.f - z) * hc[nt][r] + z * ht;
        hc[nt][r] = hv;
        if (t + 1 == sl[r])
          clinic[(size_t)(b0 + r + 8 * half) * NH + nt * 16 + ncol] = hv;
      }
    }
  }
}

// ---------------------------------------------------------------------------
// Kernel 2: GRAM attention embedding (one block of 128 threads per row)
// ---------------------------------------------------------------------------
__global__ __launch_bounds__(128) void gram_attn_kernel(
    const int* __restrict__ lv0, const int* __restrict__ an0,
    const int* __restrict__ lv1, const int* __restrict__ an1,
    const int* __restrict__ lv2, const int* __restrict__ an2,
    const float* __restrict__ ini, const float* __restrict__ W_Wa,
    const float* __restrict__ b_Wa, const float* __restrict__ W_Ua,
    float* __restrict__ embedAll) {
  const int s = blockIdx.x;
  const int e = threadIdx.x;
  const int *lv, *an;
  int L, r;
  if (s < 2000)      { lv = lv0; an = an0; L = 3; r = s; }
  else if (s < 5000) { lv = lv1; an = an1; L = 4; r = s - 2000; }
  else               { lv = lv2; an = an2; L = 5; r = s - 5000; }

  __shared__ float red[128];
  __shared__ float u[5];

  for (int l = 0; l < L; ++l) {
    const float* le = ini + (size_t)lv[r * L + l] * NE;
    const float* ae = ini + (size_t)an[r * L + l] * NE;
    float acc = b_Wa[e];
    for (int k = 0; k < NE; ++k) acc += le[k] * W_Wa[(size_t)k * NE + e];
    for (int k = 0; k < NE; ++k) acc += ae[k] * W_Wa[(size_t)(NE + k) * NE + e];
    red[e] = tanhf(acc) * W_Ua[e];
    __syncthreads();
    for (int off = 64; off > 0; off >>= 1) {
      if (e < off) red[e] += red[e + off];
      __syncthreads();
    }
    if (e == 0) u[l] = red[0];
    __syncthreads();
  }
  float mx = u[0];
  for (int l = 1; l < L; ++l) mx = fmaxf(mx, u[l]);
  float den = 0.f;
  for (int l = 0; l < L; ++l) den += __expf(u[l] - mx);
  float out = 0.f;
  for (int l = 0; l < L; ++l)
    out += (__expf(u[l] - mx) / den) * ini[(size_t)an[r * L + l] * NE + e];
  embedAll[(size_t)s * NE + e] = out;
}

// ---------------------------------------------------------------------------
// Kernel 3: code_embd = medical_code @ embedG via WMMA; A-tiles via
// GLOBAL_LOAD_ASYNC_TO_LDS double buffering; B from packed fragments.
// ---------------------------------------------------------------------------
__global__ __launch_bounds__(128) void code_gemm_kernel(
    const float* __restrict__ mc, const _Float16* __restrict__ pEmb,
    float* __restrict__ codeRaw) {
  __shared__ float astage[4][2][16][32];  // 16 KB

  const int wave = threadIdx.x >> 5;
  const int lane = threadIdx.x & 31;
  const int half = lane >> 4;
  const int ncol = lane & 15;
  const int b0   = (blockIdx.x * 4 + wave) * 16;

  v8f zero = {};
  v8f acc[8];
#pragma unroll
  for (int nt = 0; nt < 8; ++nt) acc[nt] = zero;

  const float* abase = mc + (size_t)b0 * NCODES;
  async_tile_load<32>(abase, NCODES, &astage[wave][0][0][0]);

  for (int ks = 0; ks < NCODES / 32; ++ks) {
    if (ks + 1 < NCODES / 32) {
      async_tile_load<32>(abase + (ks + 1) * 32, NCODES,
                          &astage[wave][(ks + 1) & 1][0][0]);
      wait_async_le4();
    } else {
      wait_async_0();
    }
    v16h a = afrag_f32(&astage[wave][ks & 1][0][0], 32, 0);
#pragma unroll
    for (int nt = 0; nt < 8; ++nt)
      acc[nt] = wmma_f16(a, bfrag_packed(pEmb, ks * 8 + nt), acc[nt]);
  }
#pragma unroll
  for (int nt = 0; nt < 8; ++nt)
#pragma unroll
    for (int r = 0; r < 8; ++r)
      codeRaw[(size_t)(b0 + r + 8 * half) * NE + nt * 16 + ncol] = acc[nt][r];
}

// ---------------------------------------------------------------------------
// Kernel 4: tails
// ---------------------------------------------------------------------------
__global__ __launch_bounds__(256) void tail_kernel(
    const float* __restrict__ clinic, const float* __restrict__ codeRaw,
    const float* __restrict__ W_ct, const float* __restrict__ b_ct,
    const float* __restrict__ W_cc, const float* __restrict__ b_cc,
    const float* __restrict__ W_codc, const float* __restrict__ b_codc,
    const float* __restrict__ W_cls, const float* __restrict__ b_cls,
    float* __restrict__ out) {
  const int w = threadIdx.x >> 5;
  const int j = threadIdx.x & 31;
  const int b = blockIdx.x * 8 + w;

  __shared__ float sc32[8][32];
  __shared__ float scl[8][32];

  float c32 = b_ct[j];
  const float* crow = codeRaw + (size_t)b * NE;
  for (int e = 0; e < NE; ++e) c32 += crow[e] * W_ct[e * 32 + j];
  sc32[w][j] = c32;
  scl[w][j]  = clinic[(size_t)b * NH + j];
  __syncthreads();

  float cn = b_cc[j], con = b_codc[j];
  for (int i = 0; i < 32; ++i) {
    cn  += tanhf(scl[w][i]) * W_cc[i * 32 + j];
    con += tanhf(sc32[w][i]) * W_codc[i * 32 + j];
  }
  out[NB + (size_t)b * 32 + j]           = cn;
  out[NB + NB * 32 + (size_t)b * 32 + j] = con;

  if (j == 0) {
    float p = b_cls[0];
    for (int i = 0; i < 32; ++i) p += scl[w][i] * W_cls[i];
    for (int i = 0; i < 32; ++i) p += sc32[w][i] * W_cls[32 + i];
    out[b] = 1.f / (1.f + __expf(-p));
  }
}

// ---------------------------------------------------------------------------
extern "C" void kernel_launch(void* const* d_in, const int* in_sizes, int n_in,
                              void* d_out, int out_size, void* d_ws,
                              size_t ws_size, hipStream_t stream) {
  const float* X      = (const float*)d_in[0];
  const float* Mask   = (const float*)d_in[1];
  const float* Delta  = (const float*)d_in[2];
  const float* Xlast  = (const float*)d_in[3];
  const int*   seqlen = (const int*)d_in[4];
  const float* mc     = (const float*)d_in[5];
  // d_in[6] profiles: unused by reference
  const int* lv0 = (const int*)d_in[7];
  const int* an0 = (const int*)d_in[8];
  const int* lv1 = (const int*)d_in[9];
  const int* an1 = (const int*)d_in[10];
  const int* lv2 = (const int*)d_in[11];
  const int* an2 = (const int*)d_in[12];
  const int* mapInfo = (const int*)d_in[13];
  const float* X_mean = (const float*)d_in[14];
  const float* W_z = (const float*)d_in[15]; const float* b_z = (const float*)d_in[16];
  const float* W_r = (const float*)d_in[17]; const float* b_r = (const float*)d_in[18];
  const float* W_h = (const float*)d_in[19]; const float* b_h = (const float*)d_in[20];
  const float* W_gx = (const float*)d_in[21]; const float* b_gx = (const float*)d_in[22];
  const float* W_gh = (const float*)d_in[23]; const float* b_gh = (const float*)d_in[24];
  const float* W_cls = (const float*)d_in[25]; const float* b_cls = (const float*)d_in[26];
  const float* ini = (const float*)d_in[27];
  const float* W_Wa = (const float*)d_in[28]; const float* b_Wa = (const float*)d_in[29];
  const float* W_Ua = (const float*)d_in[30];
  const float* W_ct = (const float*)d_in[31]; const float* b_ct = (const float*)d_in[32];
  const float* W_cc = (const float*)d_in[33]; const float* b_cc = (const float*)d_in[34];
  const float* W_codc = (const float*)d_in[35]; const float* b_codc = (const float*)d_in[36];

  char* ws = (char*)d_ws;  // all offsets 512B-aligned
  float*    clinic   = (float*)(ws);            // 4096*32*4   = 524288
  float*    embedAll = (float*)(ws + 524288);   // 8000*128*4  = 4096000
  float*    codeRaw  = (float*)(ws + 4620288);  // 4096*128*4  = 2097152
  _Float16* pW_gx    = (_Float16*)(ws + 6717440);  // 32  frags * 1024B
  _Float16* pW_gh    = (_Float16*)(ws + 6750208);  // 8   frags
  _Float16* pW_z     = (_Float16*)(ws + 6758400);  // 18  frags
  _Float16* pW_r     = (_Float16*)(ws + 6776832);  // 18  frags
  _Float16* pW_h     = (_Float16*)(ws + 6795264);  // 18  frags
  _Float16* pEmb     = (_Float16*)(ws + 6813696);  // 2000 frags = 2048000
  float*    out      = (float*)d_out;

  pack_weight_kernel<<<32, 32, 0, stream>>>(W_gx, NIN, 8, pW_gx);
  pack_weight_kernel<<<8,  32, 0, stream>>>(W_gh, NH, 2, pW_gh);
  pack_weight_kernel<<<18, 32, 0, stream>>>(W_z,  NH, 2, pW_z);
  pack_weight_kernel<<<18, 32, 0, stream>>>(W_r,  NH, 2, pW_r);
  pack_weight_kernel<<<18, 32, 0, stream>>>(W_h,  NH, 2, pW_h);

  grud_scan_kernel<<<NB / 64, 128, 0, stream>>>(
      X, Mask, Delta, Xlast, seqlen, X_mean, b_z, b_r, b_h, b_gx, b_gh,
      pW_z, pW_r, pW_h, pW_gx, pW_gh, clinic);

  gram_attn_kernel<<<NCODES, 128, 0, stream>>>(lv0, an0, lv1, an1, lv2, an2,
                                               ini, W_Wa, b_Wa, W_Ua, embedAll);
  pack_embed_kernel<<<2000, 32, 0, stream>>>(embedAll, mapInfo, pEmb);
  code_gemm_kernel<<<NB / 64, 128, 0, stream>>>(mc, pEmb, codeRaw);
  tail_kernel<<<NB / 8, 256, 0, stream>>>(clinic, codeRaw, W_ct, b_ct, W_cc,
                                          b_cc, W_codc, b_codc, W_cls, b_cls,
                                          out);
}
